// MOGConv_3126736192095
// MI455X (gfx1250) — matrix operand order repaired and checked by reading, not compiled, not run on hardware
//
#include <hip/hip_runtime.h>
#include <hip/hip_bf16.h>
#include <stdint.h>

typedef _Float16 f16;
typedef __attribute__((ext_vector_type(16))) _Float16 v16h;
typedef __attribute__((ext_vector_type(8)))  float    v8f;
typedef unsigned int u32x4 __attribute__((ext_vector_type(4)));
typedef int          i32x4 __attribute__((ext_vector_type(4)));
typedef int          i32x8 __attribute__((ext_vector_type(8)));

union V16H { v16h v; f16 h[16]; };
union V8F  { v8f v; float f[8]; };

__device__ __forceinline__ v8f wmma32f16(v16h a, v16h b, v8f c) {
  // D = A(16x32 f16) x B(32x16 f16) + C(16x16 f32)
  return __builtin_amdgcn_wmma_f32_16x16x32_f16(false, a, false, b, (short)0, c,
                                                false, false);
}

// A-matrix (16x32 f16) K index for VGPR pair v (0..7); hi = lane>=16.
// ISA 7.12.2: lanes0-15 V0..3 -> K=0..7, V4..7 -> K=16..23; lanes16-31 +8.
__device__ __forceinline__ int kmapA(int v, int hi) {
  return (v < 4 ? 2 * v : 8 + 2 * v) + (hi ? 8 : 0);
}
// B-matrix (32x16 f16): same K order as A (lanes0-15 V0..3 -> K=0..7, V4..7 -> K=16..23; lanes16-31 +8).
__device__ __forceinline__ int kmapB(int v, int hi) {
  return (v < 4 ? 2 * v : 8 + 2 * v) + (hi ? 8 : 0);
}

__device__ __forceinline__ float leaky(float v) { return v > 0.f ? v : 0.2f * v; }

// ------------------------------------------------------------------- TDM
// Issue a TENSOR_LOAD_TO_LDS of a 2D tile: nrows rows of rowlen f16 elements,
// row stride = rowlen (dense point-major operand), into LDS at lds_off.
// D# per CDNA5 ISA. Tracked by TENSORcnt.
__device__ __forceinline__ void tdm_load_tile_f16(uint32_t lds_off,
                                                  const void* gptr,
                                                  uint32_t rowlen,
                                                  uint32_t nrows,
                                                  uint32_t tensor_rows) {
  uint64_t ga = (uint64_t)(uintptr_t)gptr;
  uint32_t ga_lo = (uint32_t)ga;
  uint32_t ga_hi = (uint32_t)(ga >> 32);
  // force wave-uniform dwords (descriptor groups live in SGPRs)
  lds_off = (uint32_t)__builtin_amdgcn_readfirstlane((int)lds_off);
  ga_lo   = (uint32_t)__builtin_amdgcn_readfirstlane((int)ga_lo);
  ga_hi   = (uint32_t)__builtin_amdgcn_readfirstlane((int)ga_hi);

  u32x4 g0;
  g0[0] = 1u;                                        // count=1 (valid), user
  g0[1] = lds_off;                                   // lds_addr [63:32]
  g0[2] = ga_lo;                                     // global_addr [95:64]
  g0[3] = (ga_hi & 0x01FFFFFFu) | (2u << 30);        // addr[56:32] | type=2

  i32x8 g1;
  g1[0] = (int)(1u << 16);                           // data_size=1 (2 bytes)
  g1[1] = (int)((rowlen & 0xFFFFu) << 16);           // tensor_dim0[15:0]
  g1[2] = (int)(((rowlen >> 16) & 0xFFFFu) |         // tensor_dim0[31:16]
                ((tensor_rows & 0xFFFFu) << 16));    // tensor_dim1[15:0]
  g1[3] = (int)(((tensor_rows >> 16) & 0xFFFFu) |    // tensor_dim1[31:16]
                ((rowlen & 0xFFFFu) << 16));         // tile_dim0
  g1[4] = (int)(nrows & 0xFFFFu);                    // tile_dim1 (tile_dim2=0)
  g1[5] = (int)rowlen;                               // tensor_dim0_stride lo
  g1[6] = 0;
  g1[7] = 0;

  i32x4 z4 = {0, 0, 0, 0};
#if __has_include(<hip/amd_detail/amd_gfx1250_TDM.h>)
  i32x8 z8 = {0, 0, 0, 0, 0, 0, 0, 0};
  __builtin_amdgcn_tensor_load_to_lds(g0, g1, z4, z4, z8, 0);
#else
  __builtin_amdgcn_tensor_load_to_lds(g0, g1, z4, z4, 0);
#endif
}

// ---------------------------------------------------------------- elementwise
__global__ void k_zero(float* p, int cnt) {
  for (int i = threadIdx.x + blockIdx.x * blockDim.x; i < cnt;
       i += blockDim.x * gridDim.x)
    p[i] = 0.f;
}

__global__ void k_cast(const float* __restrict__ x, long total, f16* __restrict__ o) {
  long i = (long)blockIdx.x * blockDim.x + threadIdx.x;
  if (i < total) o[i] = (f16)x[i];
}

__global__ void k_pad128(const float* __restrict__ x, int n, int c,
                         f16* __restrict__ xp) {
  long i = (long)blockIdx.x * blockDim.x + threadIdx.x;
  if (i >= (long)n * 128) return;
  int p = (int)(i >> 7), ch = (int)(i & 127);
  xp[i] = (ch < c) ? (f16)x[(long)p * c + ch] : (f16)0.f;
}

__global__ void k_sqnorm(const float* __restrict__ x, int n, int c,
                         float* __restrict__ sq) {
  int i = blockIdx.x * blockDim.x + threadIdx.x;
  if (i >= n) return;
  float s = 0.f;
  for (int j = 0; j < c; ++j) { float v = x[(long)i * c + j]; s += v * v; }
  sq[i] = s;
}

// ------------------------------------------------------------------- KNN
// One block = 16 query rows. Phase 1: 8 waves compute X_q @ X_c^T in 16x16
// tiles via WMMA, convert to squared distance, mask, stash f16 strip in LDS
// (16 x n x 2B = 256KB, fits the 320KB WGP LDS). Phase 2: per-row iterative
// argmin x20 with wave32 shuffle reduction.
__global__ __launch_bounds__(256) void k_knn(const f16* __restrict__ xpad,
                                             const float* __restrict__ sq,
                                             const int* __restrict__ tag,
                                             int n, int* __restrict__ nbr) {
  extern __shared__ f16 sdist[];
  const int lane = threadIdx.x & 31;
  const int wave = threadIdx.x >> 5;
  const int q0 = blockIdx.x * 16;
  const int nl = lane & 15;
  const int hi = lane >> 4;

  V16H Areg[4];
  const int ar = q0 + nl;
  for (int kc = 0; kc < 4; ++kc)
    for (int v = 0; v < 8; ++v) {
      int k = kc * 32 + kmapA(v, hi);
      Areg[kc].h[2 * v]     = xpad[(long)ar * 128 + k];
      Areg[kc].h[2 * v + 1] = xpad[(long)ar * 128 + k + 1];
    }
  float sqq[8]; int tq[8];
  for (int r = 0; r < 8; ++r) {
    int mr = q0 + (hi ? 8 : 0) + r;
    sqq[r] = sq[mr]; tq[r] = tag[mr];
  }

  for (int ct = wave; ct < (n >> 4); ct += 8) {
    const int c0 = ct * 16;
    V8F acc;
    for (int i = 0; i < 8; ++i) acc.f[i] = 0.f;
    for (int kc = 0; kc < 4; ++kc) {
      V16H bv;
      for (int v = 0; v < 8; ++v) {
        int k = kc * 32 + kmapB(v, hi);
        bv.h[2 * v]     = xpad[(long)(c0 + nl) * 128 + k];
        bv.h[2 * v + 1] = xpad[(long)(c0 + nl) * 128 + k + 1];
      }
      acc.v = wmma32f16(Areg[kc].v, bv.v, acc.v);
    }
    float sqc = sq[c0 + nl];
    int   tc  = tag[c0 + nl];
    for (int r = 0; r < 8; ++r) {
      int mr = (hi ? 8 : 0) + r;
      int qp = q0 + mr, cp = c0 + nl;
      float d = sqq[r] + sqc - 2.0f * acc.f[r];
      if (tq[r] != tc || qp == cp) d = __builtin_inff();
      sdist[mr * n + cp] = (f16)d;
    }
  }
  __syncthreads();

  for (int m = wave * 2; m < wave * 2 + 2; ++m) {
    f16* row = sdist + m * n;
    for (int t = 0; t < 20; ++t) {
      float best = __builtin_inff();
      int   bi   = 0x7fffffff;
      for (int j = lane; j < n; j += 32) {
        float v = (float)row[j];
        if (v < best) { best = v; bi = j; }
      }
      for (int msk = 16; msk >= 1; msk >>= 1) {
        float ov = __shfl_xor(best, msk, 32);
        int   oi = __shfl_xor(bi, msk, 32);
        if (ov < best || (ov == best && oi < bi)) { best = ov; bi = oi; }
      }
      int sel = (bi < n) ? bi : 0;
      row[sel] = (f16)__builtin_inff();     // all lanes: benign same-value store
      if (lane == 0) nbr[(long)(q0 + m) * 20 + t] = sel;
    }
  }
}

// ------------------------------------------------------------- gather GEMM
// Y(MxE) = A(MxKd f32->f16) x B(KdxE f16), B gathered per the faithful
// x[idx].reshape(c, e): flat = k*E+col; j = flat/cA; ch = flat%cA.
// Fused per-row sum/sumsq for InstanceNorm.
__global__ __launch_bounds__(256) void k_gemm_gather(
    const float* __restrict__ A, int M, int Kd,
    const f16* __restrict__ Bsrc, const int* __restrict__ idx, int cA, long E,
    float* __restrict__ Y, float* __restrict__ sum, float* __restrict__ sumsq) {
  const int lane = threadIdx.x & 31;
  const int wave = threadIdx.x >> 5;
  const int mtiles = M >> 4;
  const long ntiles = E >> 4;
  long tile = (long)blockIdx.x * 8 + wave;
  if (tile >= (long)mtiles * ntiles) return;
  const int  mt  = (int)(tile % mtiles);
  const long nt  = tile / mtiles;
  const int  m0  = mt << 4;
  const int  nl  = lane & 15;
  const int  hi  = lane >> 4;
  const long col = (nt << 4) + nl;

  V8F acc;
  for (int i = 0; i < 8; ++i) acc.f[i] = 0.f;

  const int arow = m0 + nl;
  const int kcn = (Kd + 31) >> 5;
  for (int kc = 0; kc < kcn; ++kc) {
    const int kb = kc << 5;
    V16H av, bv;
    for (int v = 0; v < 8; ++v) {
      int ka = kb + kmapA(v, hi);
      av.h[2*v]   = (ka     < Kd) ? (f16)A[(long)arow * Kd + ka]     : (f16)0.f;
      av.h[2*v+1] = (ka + 1 < Kd) ? (f16)A[(long)arow * Kd + ka + 1] : (f16)0.f;
      int kbb = kb + kmapB(v, hi);
      f16 b0 = (f16)0.f, b1 = (f16)0.f;
      if (kbb < Kd) {
        long flat = (long)kbb * E + col;
        b0 = Bsrc[(long)idx[flat / cA] * cA + (int)(flat % cA)];
      }
      if (kbb + 1 < Kd) {
        long flat = (long)(kbb + 1) * E + col;
        b1 = Bsrc[(long)idx[flat / cA] * cA + (int)(flat % cA)];
      }
      bv.h[2*v] = b0; bv.h[2*v+1] = b1;
    }
    acc.v = wmma32f16(av.v, bv.v, acc.v);
  }

  for (int r = 0; r < 8; ++r) {
    int mr = m0 + (hi ? 8 : 0) + r;
    Y[(long)mr * E + col] = acc.f[r];
  }
  if (sum != nullptr) {
    for (int r = 0; r < 8; ++r) {
      float s = acc.f[r], s2 = s * s;
      for (int msk = 8; msk >= 1; msk >>= 1) {   // reduce within 16-lane group
        s  += __shfl_xor(s,  msk, 32);
        s2 += __shfl_xor(s2, msk, 32);
      }
      if (nl == 0) {
        int mr = m0 + (hi ? 8 : 0) + r;
        atomicAdd(&sum[mr], s);
        atomicAdd(&sumsq[mr], s2);
      }
    }
  }
}

// --------------------------------------------------------------- TDM GEMM
// Dense point-major B operand (Bsrc[col*Kd + k]). Each wave DMAs its
// 16 x Kd f16 B-tile into a private LDS slot via the Tensor Data Mover,
// waits on TENSORcnt, then runs the WMMA k-loop from LDS.
__global__ __launch_bounds__(256) void k_gemm_tdm(
    const float* __restrict__ A, int M, int Kd,
    const f16* __restrict__ Bsrc, long E,
    float* __restrict__ Y, float* __restrict__ sum, float* __restrict__ sumsq) {
  extern __shared__ f16 sB[];
  const int lane = threadIdx.x & 31;
  const int wave = __builtin_amdgcn_readfirstlane((int)(threadIdx.x >> 5));
  const int mtiles = M >> 4;
  const long ntiles = E >> 4;
  long tile = (long)blockIdx.x * 8 + wave;
  if (tile >= (long)mtiles * ntiles) return;
  const int  mt  = (int)(tile % mtiles);
  const long nt  = tile / mtiles;
  const int  m0  = mt << 4;
  const int  nl  = lane & 15;
  const int  hi  = lane >> 4;
  const long col0 = nt << 4;

  const uint32_t slotB = (uint32_t)wave * 16u * (uint32_t)Kd * 2u;
  const uint32_t ldsBase = (uint32_t)(uintptr_t)(&sB[0]);
  tdm_load_tile_f16(ldsBase + slotB, Bsrc + col0 * (long)Kd,
                    (uint32_t)Kd, 16u, (uint32_t)E);

  // acc init while TDM streams B; A (f32 weights -> f16) loads after the wait
  V8F acc;
  for (int i = 0; i < 8; ++i) acc.f[i] = 0.f;
  const int arow = m0 + nl;
  const int kcn = (Kd + 31) >> 5;

  __builtin_amdgcn_s_wait_tensorcnt(0);
  const f16* bt = sB + (size_t)wave * 16 * Kd + (size_t)nl * Kd;

  for (int kc = 0; kc < kcn; ++kc) {
    const int kb = kc << 5;
    V16H av, bv;
    for (int v = 0; v < 8; ++v) {
      int ka = kb + kmapA(v, hi);
      av.h[2*v]   = (ka     < Kd) ? (f16)A[(long)arow * Kd + ka]     : (f16)0.f;
      av.h[2*v+1] = (ka + 1 < Kd) ? (f16)A[(long)arow * Kd + ka + 1] : (f16)0.f;
      int kbb = kb + kmapB(v, hi);
      bv.h[2*v]   = (kbb     < Kd) ? bt[kbb]     : (f16)0.f;
      bv.h[2*v+1] = (kbb + 1 < Kd) ? bt[kbb + 1] : (f16)0.f;
    }
    acc.v = wmma32f16(av.v, bv.v, acc.v);
  }

  for (int r = 0; r < 8; ++r) {
    int mr = m0 + (hi ? 8 : 0) + r;
    Y[(long)mr * E + col0 + nl] = acc.f[r];
  }
  if (sum != nullptr) {
    for (int r = 0; r < 8; ++r) {
      float s = acc.f[r], s2 = s * s;
      for (int msk = 8; msk >= 1; msk >>= 1) {
        s  += __shfl_xor(s,  msk, 32);
        s2 += __shfl_xor(s2, msk, 32);
      }
      if (nl == 0) {
        int mr = m0 + (hi ? 8 : 0) + r;
        atomicAdd(&sum[mr], s);
        atomicAdd(&sumsq[mr], s2);
      }
    }
  }
}

// -------------------------------------------------------- norm / act / misc
__global__ void k_norm_act_T(const float* __restrict__ Y,
                             const float* __restrict__ sum,
                             const float* __restrict__ sumsq,
                             int M, long E, f16* __restrict__ BT) {
  long i = (long)blockIdx.x * blockDim.x + threadIdx.x;
  if (i >= (long)M * E) return;
  int  m   = (int)(i / E);
  long col = i % E;
  float mu = sum[m] / (float)E;
  float rs = rsqrtf(sumsq[m] / (float)E - mu * mu + 1e-5f);
  float v  = leaky((Y[i] - mu) * rs);
  BT[col * M + m] = (f16)v;
}

__global__ void k_init_agg(uint32_t* p, long total) {
  long i = (long)blockIdx.x * blockDim.x + threadIdx.x;
  if (i < total) p[i] = 0x007FFFFFu;       // encode(-inf)
}

__global__ void k_norm_act_scatter(const float* __restrict__ Y,
                                   const float* __restrict__ sum,
                                   const float* __restrict__ sumsq,
                                   int M, long E, const int* __restrict__ idx,
                                   uint32_t* __restrict__ agg) {
  long i = (long)blockIdx.x * blockDim.x + threadIdx.x;
  if (i >= (long)M * E) return;
  int  m   = (int)(i / E);
  long col = i % E;
  float mu = sum[m] / (float)E;
  float rs = rsqrtf(sumsq[m] / (float)E - mu * mu + 1e-5f);
  float v  = leaky((Y[i] - mu) * rs);
  uint32_t b = __float_as_uint(v);
  uint32_t e = (b & 0x80000000u) ? ~b : (b | 0x80000000u);  // order-preserving
  atomicMax(&agg[(long)idx[col] * M + m], e);
}

__global__ void k_fixup(const uint32_t* __restrict__ agg,
                        const float* __restrict__ resid,
                        float* __restrict__ xo, f16* __restrict__ xh, long total) {
  long i = (long)blockIdx.x * blockDim.x + threadIdx.x;
  if (i >= total) return;
  uint32_t e = agg[i];
  uint32_t b = (e & 0x80000000u) ? (e & 0x7fffffffu) : ~e;
  float v = __uint_as_float(b);
  if (v == -__builtin_inff()) v = 0.f;      // empty segment -> 0
  if (resid) v += resid[i];
  xo[i] = v;
  xh[i] = (f16)v;
}

__global__ void k_concat4(const f16* a, const f16* b, const f16* c, const f16* d,
                          int n, f16* __restrict__ o) {
  long i = (long)blockIdx.x * blockDim.x + threadIdx.x;
  if (i >= (long)n * 512) return;
  int p = (int)(i >> 9), k = (int)(i & 511);
  const f16* s = (k < 128) ? a : (k < 256) ? b : (k < 384) ? c : d;
  o[i] = s[(long)p * 128 + (k & 127)];
}

__global__ void k_rowmax(const float* __restrict__ Y, const float* __restrict__ sum,
                         const float* __restrict__ sumsq, long E,
                         float* __restrict__ g) {
  int m = blockIdx.x;
  float mu = sum[m] / (float)E;
  float rs = rsqrtf(sumsq[m] / (float)E - mu * mu + 1e-5f);
  float best = -__builtin_inff();
  for (long j = threadIdx.x; j < E; j += blockDim.x)
    best = fmaxf(best, leaky((Y[(long)m * E + j] - mu) * rs));
  __shared__ float red[256];
  red[threadIdx.x] = best;
  __syncthreads();
  for (int s = 128; s > 0; s >>= 1) {
    if (threadIdx.x < s) red[threadIdx.x] = fmaxf(red[threadIdx.x], red[threadIdx.x + s]);
    __syncthreads();
  }
  if (threadIdx.x == 0) g[m] = red[0];
}

__global__ void k_concat_g(const float* __restrict__ g, const f16* __restrict__ xh,
                           int n, f16* __restrict__ o) {
  long i = (long)blockIdx.x * blockDim.x + threadIdx.x;
  if (i >= (long)n * 384) return;
  int p = (int)(i / 384), k = (int)(i % 384);
  o[i] = (k < 256) ? (f16)g[k] : xh[(long)p * 128 + (k - 256)];
}

__global__ void k_concat_yx(const f16* __restrict__ y, int My,
                            const f16* __restrict__ xh, int n, int Kt,
                            f16* __restrict__ o) {
  long i = (long)blockIdx.x * blockDim.x + threadIdx.x;
  if (i >= (long)n * Kt) return;
  int p = (int)(i / Kt), k = (int)(i % Kt);
  o[i] = (k < My) ? y[(long)p * My + k] : xh[(long)p * 128 + (k - My)];
}

__global__ void k_final(const float* __restrict__ Yc, const float* __restrict__ br,
                        const float* __restrict__ pc, int n, float* __restrict__ out) {
  long i = (long)blockIdx.x * blockDim.x + threadIdx.x;
  long nc = (long)n * 128;
  if (i < nc) {
    int p = (int)(i >> 7), o = (int)(i & 127);
    out[i] = Yc[(long)o * n + p] + br[o];
  } else if (i < nc + (long)n * 3) {
    out[i] = pc[i - nc];
  }
}

// ------------------------------------------------------------------- driver
extern "C" void kernel_launch(void* const* d_in, const int* in_sizes, int n_in,
                              void* d_out, int out_size, void* d_ws, size_t ws_size,
                              hipStream_t stream) {
  const float* pc  = (const float*)d_in[0];
  const int*   tag = (const int*)  d_in[1];
  const float* Wb_[13];
  for (int i = 0; i < 13; ++i) Wb_[i] = (const float*)d_in[2 + i];
  const float* Wr = (const float*)d_in[15];
  const float* br = (const float*)d_in[16];
  float* out = (float*)d_out;

  const int  n  = in_sizes[0] / 3;     // 8192
  const long E  = (long)n * 20;        // 163840 edges

  char*  ws  = (char*)d_ws;
  size_t off = 0;
  auto alloc = [&](size_t bytes) -> void* {
    void* p = ws + off;
    off = (off + bytes + 255) & ~(size_t)255;
    return p;
  };
  float*    sq    = (float*)alloc((size_t)n * 4);
  int*      nbr   = (int*)alloc((size_t)n * 20 * 4);
  f16*      x0h   = (f16*)alloc((size_t)n * 3 * 2);
  f16*      xpad0 = (f16*)alloc((size_t)n * 128 * 2);
  float*    x[4];
  f16*      xh[4];
  for (int i = 0; i < 4; ++i) x[i]  = (float*)alloc((size_t)n * 128 * 4);
  for (int i = 0; i < 4; ++i) xh[i] = (f16*)alloc((size_t)n * 128 * 2);
  float*    Ybuf  = (float*)alloc((size_t)128 * E * 4);
  f16*      BT    = (f16*)alloc((size_t)E * 128 * 2);
  uint32_t* agg   = (uint32_t*)alloc((size_t)n * 128 * 4);
  float*    stats = (float*)alloc(1024 * 4);
  float*    sum   = stats, *sumsq = stats + 512;
  float*    g     = (float*)alloc(256 * 4);
  f16*      cat   = (f16*)alloc((size_t)n * 512 * 2);
  f16*      yA    = (f16*)alloc((size_t)n * 256 * 2);
  f16*      yB    = (f16*)alloc((size_t)n * 256 * 2);

  auto gc = [](long t) { return (unsigned)((t + 255) / 256); };
  auto gemmG = [&](const float* A, int M, int Kd, const f16* B, const int* idx,
                   int cA, long Ecols, float* Y, float* sm, float* s2) {
    long tiles = (long)(M >> 4) * (Ecols >> 4);
    k_gemm_gather<<<(unsigned)((tiles + 7) / 8), 256, 0, stream>>>(
        A, M, Kd, B, idx, cA, Ecols, Y, sm, s2);
  };
  auto gemmT = [&](const float* A, int M, int Kd, const f16* B, long Ecols,
                   float* Y, float* sm, float* s2) {
    long tiles = (long)(M >> 4) * (Ecols >> 4);
    size_t shmem = (size_t)8 * 16 * (size_t)Kd * sizeof(f16);  // 8 wave slots
    k_gemm_tdm<<<(unsigned)((tiles + 7) / 8), 256, shmem, stream>>>(
        A, M, Kd, B, Ecols, Y, sm, s2);
  };

  // layer-0 operand prep
  k_cast<<<gc((long)n * 3), 256, 0, stream>>>(pc, (long)n * 3, x0h);
  k_pad128<<<gc((long)n * 128), 256, 0, stream>>>(pc, n, 3, xpad0);

  auto edge_block = [&](const float* xin, int cin, const f16* xinh,
                        const f16* xpad, const float* Wa, const float* Wbm,
                        const float* resid, float* xo, f16* xoh) {
    k_sqnorm<<<gc(n), 256, 0, stream>>>(xin, n, cin, sq);
    k_knn<<<n / 16, 256, (size_t)16 * n * sizeof(f16), stream>>>(xpad, sq, tag, n, nbr);
    k_zero<<<4, 256, 0, stream>>>(stats, 1024);
    gemmG(Wa, 128, cin, xinh, nbr, cin, E, Ybuf, sum, sumsq);
    k_norm_act_T<<<gc(128 * E), 256, 0, stream>>>(Ybuf, sum, sumsq, 128, E, BT);
    k_zero<<<4, 256, 0, stream>>>(stats, 1024);
    gemmT(Wbm, 128, 128, BT, E, Ybuf, sum, sumsq);
    k_init_agg<<<gc((long)n * 128), 256, 0, stream>>>(agg, (long)n * 128);
    k_norm_act_scatter<<<gc(128 * E), 256, 0, stream>>>(Ybuf, sum, sumsq, 128, E,
                                                        nbr, agg);
    k_fixup<<<gc((long)n * 128), 256, 0, stream>>>(agg, resid, xo, xoh,
                                                   (long)n * 128);
  };

  edge_block(pc,   3,   x0h,   xpad0, Wb_[0], Wb_[1], nullptr, x[0], xh[0]);
  edge_block(x[0], 128, xh[0], xh[0], Wb_[2], Wb_[3], x[0],    x[1], xh[1]);
  edge_block(x[1], 128, xh[1], xh[1], Wb_[4], Wb_[5], x[1],    x[2], xh[2]);
  edge_block(x[2], 128, xh[2], xh[2], Wb_[6], Wb_[7], x[2],    x[3], xh[3]);

  // decoder
  k_concat4<<<gc((long)n * 512), 256, 0, stream>>>(xh[0], xh[1], xh[2], xh[3], n, cat);
  k_zero<<<4, 256, 0, stream>>>(stats, 1024);
  gemmT(Wb_[8], 256, 512, cat, (long)n, Ybuf, sum, sumsq);
  k_rowmax<<<256, 256, 0, stream>>>(Ybuf, sum, sumsq, (long)n, g);

  k_concat_g<<<gc((long)n * 384), 256, 0, stream>>>(g, xh[3], n, cat);
  k_zero<<<4, 256, 0, stream>>>(stats, 1024);
  gemmT(Wb_[9], 256, 384, cat, (long)n, Ybuf, sum, sumsq);
  k_norm_act_T<<<gc((long)256 * n), 256, 0, stream>>>(Ybuf, sum, sumsq, 256, (long)n, yA);

  k_concat_yx<<<gc((long)n * 384), 256, 0, stream>>>(yA, 256, xh[2], n, 384, cat);
  k_zero<<<4, 256, 0, stream>>>(stats, 1024);
  gemmT(Wb_[10], 256, 384, cat, (long)n, Ybuf, sum, sumsq);
  k_norm_act_T<<<gc((long)256 * n), 256, 0, stream>>>(Ybuf, sum, sumsq, 256, (long)n, yB);

  k_concat_yx<<<gc((long)n * 384), 256, 0, stream>>>(yB, 256, xh[1], n, 384, cat);
  k_zero<<<4, 256, 0, stream>>>(stats, 1024);
  gemmT(Wb_[11], 128, 384, cat, (long)n, Ybuf, sum, sumsq);
  k_norm_act_T<<<gc((long)128 * n), 256, 0, stream>>>(Ybuf, sum, sumsq, 128, (long)n, yA);

  k_concat_yx<<<gc((long)n * 256), 256, 0, stream>>>(yA, 128, xh[0], n, 256, cat);
  k_zero<<<4, 256, 0, stream>>>(stats, 1024);
  gemmT(Wb_[12], 128, 256, cat, (long)n, Ybuf, sum, sumsq);
  k_norm_act_T<<<gc((long)128 * n), 256, 0, stream>>>(Ybuf, sum, sumsq, 128, (long)n, yB);

  gemmT(Wr, 128, 128, yB, (long)n, Ybuf, nullptr, nullptr);
  k_final<<<gc((long)n * 131), 256, 0, stream>>>(Ybuf, br, pc, n, out);
}
